// AttentionBlock_15135464751627
// MI455X (gfx1250) — matrix operand order repaired and checked by reading, not compiled, not run on hardware
//
#include <hip/hip_runtime.h>
#include <hip/hip_bf16.h>

typedef __attribute__((ext_vector_type(16))) __bf16 v16bf;
typedef __attribute__((ext_vector_type(8)))  __bf16 v8bf;
typedef __attribute__((ext_vector_type(8)))  float  v8f;
typedef __attribute__((ext_vector_type(4)))  int    v4i;

#define AS1 __attribute__((address_space(1)))
#define AS3 __attribute__((address_space(3)))

#if __has_builtin(__builtin_amdgcn_global_load_async_to_lds_b128) && \
    __has_builtin(__builtin_amdgcn_s_wait_asynccnt)
#define USE_ASYNC_LDS 1
#else
#define USE_ASYNC_LDS 0
#endif

// 16-byte global -> LDS copy. Async (ASYNCcnt) when the toolchain exposes it.
static __device__ inline void copy16_g2l(const __bf16* g, __bf16* l) {
#if USE_ASYNC_LDS
  __builtin_amdgcn_global_load_async_to_lds_b128(
      (AS1 v4i*)(void*)g, (AS3 v4i*)(void*)l, 0, 0);
#else
  *(v8bf*)l = *(const v8bf*)g;
#endif
}

static __device__ inline void async_wait_all() {
#if USE_ASYNC_LDS
  __builtin_amdgcn_s_wait_asynccnt(0);
#endif
}

static __device__ inline v8f zero_v8f() {
  v8f z;
#pragma unroll
  for (int i = 0; i < 8; ++i) z[i] = 0.f;
  return z;
}

// A fragment (16x32 bf16, MxK) from row-major LDS tile (row stride ld elems).
// lane L: m=L&15, h=L>>4.  a[0..7]=A[m][h*8..], a[8..15]=A[m][16+h*8..]  (2x b128)
static __device__ inline v16bf frag_a(const __bf16* tile, int ld, int lane) {
  const int m = lane & 15, h = lane >> 4;
  const __bf16* p0 = tile + m * ld + h * 8;
  const __bf16* p1 = p0 + 16;
  v16bf a;
#pragma unroll
  for (int i = 0; i < 8; ++i) { a[i] = p0[i]; a[8 + i] = p1[i]; }
  return a;
}

// B fragment (32x16 bf16, KxN) from LDS tile stored TRANSPOSED BT[n][k].
// lane L: n=L&15, h=L>>4.  b[0..7]=BT[n][h*8..], b[8..15]=BT[n][16+h*8..]  (2x b128)
static __device__ inline v16bf frag_b(const __bf16* tileT, int ld, int lane) {
  const int n = lane & 15, h = lane >> 4;
  const __bf16* p0 = tileT + n * ld + h * 8;
  const __bf16* p1 = p0 + 16;
  v16bf b;
#pragma unroll
  for (int i = 0; i < 8; ++i) { b[i] = p0[i]; b[8 + i] = p1[i]; }
  return b;
}

static __device__ inline v8f wmma_bf16(v16bf a, v16bf b, v8f c) {
  return __builtin_amdgcn_wmma_f32_16x16x32_bf16(false, a, false, b, (short)0, c, false, false);
}

// ---------------------------------------------------------------------------
// K0: GroupNorm (stats over 16384 elems per (b,g)) + normalize + bf16 cast
// ---------------------------------------------------------------------------
__global__ __launch_bounds__(256) void gn_kernel(const float* __restrict__ x,
                                                 const float* __restrict__ scale,
                                                 const float* __restrict__ bias,
                                                 __bf16* __restrict__ hbf) {
  const int b = blockIdx.x >> 5;
  const int g = blockIdx.x & 31;
  const int tid = threadIdx.x;
  const float* xb = x + (size_t)b * 1024 * 512;

  float s = 0.f, ss = 0.f;
#pragma unroll
  for (int ii = 0; ii < 64; ++ii) {
    const int i = tid + ii * 256;
    const int t = i >> 4, j = i & 15;
    const float v = xb[t * 512 + g * 16 + j];
    s += v; ss += v * v;
  }
  __shared__ float rs[256], rss[256];
  rs[tid] = s; rss[tid] = ss;
  __syncthreads();
  for (int off = 128; off > 0; off >>= 1) {
    if (tid < off) { rs[tid] += rs[tid + off]; rss[tid] += rss[tid + off]; }
    __syncthreads();
  }
  const float mean = rs[0] * (1.f / 16384.f);
  const float var  = rss[0] * (1.f / 16384.f) - mean * mean;
  const float rstd = rsqrtf(var + 1e-5f);

#pragma unroll
  for (int ii = 0; ii < 64; ++ii) {
    const int i = tid + ii * 256;
    const int t = i >> 4, j = i & 15, c = g * 16 + j;
    const float v = xb[t * 512 + c];
    hbf[((size_t)b * 1024 + t) * 512 + c] =
        (__bf16)((v - mean) * rstd * scale[c] + bias[c]);
  }
}

__global__ void cast_f32_bf16(const float* __restrict__ s, __bf16* __restrict__ d, int n) {
  const int i = blockIdx.x * blockDim.x + threadIdx.x;
  if (i < n) d[i] = (__bf16)s[i];
}

// ---------------------------------------------------------------------------
// K3: QKV GEMM (8192x512)@(512x6144), 128x128 tiles, 8 waves, double-buffered.
// A staged async to LDS; B loaded to VGPRs early, ds_store'd after the WMMAs.
// A-fragments pipelined one m-tile ahead of the consuming WMMA.
// Epilogue scatters into qT[b][h][c][t], kT[b][h][d][t], v[b][h][t][d].
// ---------------------------------------------------------------------------
__global__ __launch_bounds__(256) void gemm_qkv(const __bf16* __restrict__ A,
                                                const __bf16* __restrict__ B,
                                                const float* __restrict__ bias,
                                                __bf16* __restrict__ qT,
                                                __bf16* __restrict__ kT,
                                                __bf16* __restrict__ V) {
  const int row0 = blockIdx.x * 128;
  const int col0 = blockIdx.y * 128;
  __shared__ __align__(16) __bf16 sA[2][128 * 32];
  __shared__ __align__(16) __bf16 sBT[2][128 * 32];  // [n][k]
  const int tid = threadIdx.x, lane = tid & 31, w = tid >> 5;
  const int ar = tid >> 2, ak = (tid & 3) * 8;       // A: rows ar, ar+64
  const int bk = tid >> 4, bn = (tid & 15) * 8;      // B: k rows bk, bk+16

  v8f acc[8];
#pragma unroll
  for (int i = 0; i < 8; ++i) acc[i] = zero_v8f();

  auto issue_a = [&](int buf, int k0) {
#pragma unroll
    for (int ii = 0; ii < 2; ++ii) {
      const int r = ar + ii * 64;
      copy16_g2l(&A[(size_t)(row0 + r) * 512 + k0 + ak], &sA[buf][r * 32 + ak]);
    }
  };
  auto load_b = [&](int k0, v8bf* br) {
#pragma unroll
    for (int ii = 0; ii < 2; ++ii)
      br[ii] = *(const v8bf*)&B[(size_t)(k0 + bk + ii * 16) * 6144 + col0 + bn];
  };
  auto store_b = [&](int buf, v8bf* br) {
#pragma unroll
    for (int ii = 0; ii < 2; ++ii)
#pragma unroll
      for (int e = 0; e < 8; ++e)
        sBT[buf][(bn + e) * 32 + bk + ii * 16] = br[ii][e];
  };

  { v8bf br[2]; issue_a(0, 0); load_b(0, br); store_b(0, br); }

  for (int kt = 0; kt < 16; ++kt) {
    async_wait_all();
    __syncthreads();
    const int cur = kt & 1;
    const bool more = (kt + 1) < 16;
    v8bf br[2];
    if (more) { issue_a(cur ^ 1, (kt + 1) * 32); load_b((kt + 1) * 32, br); }
    const v16bf bf = frag_b(&sBT[cur][w * 16 * 32], 32, lane);
    v16bf af = frag_a(&sA[cur][0], 32, lane);
#pragma unroll
    for (int mt = 0; mt < 8; ++mt) {
      v16bf nxt = af;
      if (mt < 7) nxt = frag_a(&sA[cur][(mt + 1) * 16 * 32], 32, lane);
      acc[mt] = wmma_bf16(af, bf, acc[mt]);
      af = nxt;
    }
    if (more) store_b(cur ^ 1, br);
  }

  const int nglob = col0 + w * 16 + (lane & 15);
  const int head = nglob / 1536, rem = nglob % 1536;
  const int cc = rem / 3, comp = rem % 3;
  const int hl = lane >> 4;
  const int b = row0 >> 10;
  const float bv = bias[nglob];
  const size_t bh = (size_t)(b * 4 + head);
#pragma unroll
  for (int mt = 0; mt < 8; ++mt) {
#pragma unroll
    for (int r = 0; r < 8; ++r) {
      const int mg = row0 + mt * 16 + 8 * hl + r;
      const int tl = mg & 1023;
      const __bf16 val = (__bf16)(acc[mt][r] + bv);
      if (comp == 0)      qT[(bh * 512 + cc) * 1024 + tl] = val;
      else if (comp == 1) kT[(bh * 512 + cc) * 1024 + tl] = val;
      else                V[(bh * 1024 + tl) * 512 + cc] = val;
    }
  }
}

// ---------------------------------------------------------------------------
// K4: fused logits + softmax per (b,h). Strip M=64 (c), N=512 (d), K=1024 (t).
// 16 waves: mt=w&3, ng=w>>2 owns 8 N-tiles. Fully async staging, double buffer.
// B-fragments pipelined one n-tile ahead of the consuming WMMA.
// ---------------------------------------------------------------------------
__global__ __launch_bounds__(512) void attn_softmax(const __bf16* __restrict__ qT,
                                                    const __bf16* __restrict__ kT,
                                                    __bf16* __restrict__ attn) {
  const int bh = blockIdx.x;
  const int r0 = blockIdx.y * 64;
  const __bf16* Q = qT + (size_t)bh * 512 * 1024;
  const __bf16* K = kT + (size_t)bh * 512 * 1024;
  __shared__ __align__(16) __bf16 sA[2][64 * 32];
  __shared__ __align__(16) __bf16 sBT[2][512 * 32];  // kT already [d][t]
  __shared__ float red[4][16][4];
  const int tid = threadIdx.x, lane = tid & 31, w = tid >> 5;
  const int mt = w & 3, ng = w >> 2, hl = lane >> 4, nn = lane & 15;
  const int qr = tid >> 2, qk = (tid & 3) * 8;       // valid for tid<256

  v8f acc[8];
#pragma unroll
  for (int i = 0; i < 8; ++i) acc[i] = zero_v8f();

  auto issue = [&](int buf, int k0) {
    if (tid < 256)
      copy16_g2l(&Q[(size_t)(r0 + qr) * 1024 + k0 + qk], &sA[buf][qr * 32 + qk]);
#pragma unroll
    for (int ii = 0; ii < 4; ++ii) {
      const int i = tid + ii * 512;
      const int d = i >> 2, kk = (i & 3) * 8;
      copy16_g2l(&K[(size_t)d * 1024 + k0 + kk], &sBT[buf][d * 32 + kk]);
    }
  };

  issue(0, 0);
  for (int kt = 0; kt < 32; ++kt) {
    async_wait_all();
    __syncthreads();
    const int cur = kt & 1;
    if (kt + 1 < 32) issue(cur ^ 1, (kt + 1) * 32);
    const v16bf af = frag_a(&sA[cur][mt * 16 * 32], 32, lane);
    v16bf bf = frag_b(&sBT[cur][(ng * 128) * 32], 32, lane);
#pragma unroll
    for (int j = 0; j < 8; ++j) {
      v16bf nxt = bf;
      if (j < 7) nxt = frag_b(&sBT[cur][(ng * 128 + (j + 1) * 16) * 32], 32, lane);
      acc[j] = wmma_bf16(af, bf, acc[j]);
      bf = nxt;
    }
  }
  __syncthreads();

  const float scale = 0.03125f;  // 1/sqrt(T) = 1/32
  float rmax[8];
#pragma unroll
  for (int r = 0; r < 8; ++r) {
    float pm = -3.0e38f;
#pragma unroll
    for (int j = 0; j < 8; ++j) pm = fmaxf(pm, acc[j][r] * scale);
    pm = fmaxf(pm, __shfl_xor(pm, 1));
    pm = fmaxf(pm, __shfl_xor(pm, 2));
    pm = fmaxf(pm, __shfl_xor(pm, 4));
    pm = fmaxf(pm, __shfl_xor(pm, 8));
    if (nn == 0) red[mt][8 * hl + r][ng] = pm;
  }
  __syncthreads();
#pragma unroll
  for (int r = 0; r < 8; ++r) {
    float m = red[mt][8 * hl + r][0];
    m = fmaxf(m, red[mt][8 * hl + r][1]);
    m = fmaxf(m, red[mt][8 * hl + r][2]);
    m = fmaxf(m, red[mt][8 * hl + r][3]);
    rmax[r] = m;
  }
  __syncthreads();
  float rsum[8];
#pragma unroll
  for (int r = 0; r < 8; ++r) {
    float ps = 0.f;
#pragma unroll
    for (int j = 0; j < 8; ++j) ps += __expf(acc[j][r] * scale - rmax[r]);
    ps += __shfl_xor(ps, 1);
    ps += __shfl_xor(ps, 2);
    ps += __shfl_xor(ps, 4);
    ps += __shfl_xor(ps, 8);
    if (nn == 0) red[mt][8 * hl + r][ng] = ps;
  }
  __syncthreads();
#pragma unroll
  for (int r = 0; r < 8; ++r)
    rsum[r] = red[mt][8 * hl + r][0] + red[mt][8 * hl + r][1] +
              red[mt][8 * hl + r][2] + red[mt][8 * hl + r][3];

  __bf16* out = attn + (size_t)bh * 512 * 512;
#pragma unroll
  for (int r = 0; r < 8; ++r) {
    const int rg = r0 + mt * 16 + 8 * hl + r;
    const float inv = 1.f / rsum[r];
#pragma unroll
    for (int j = 0; j < 8; ++j) {
      const int dg = ng * 128 + j * 16 + nn;
      out[(size_t)rg * 512 + dg] = (__bf16)(__expf(acc[j][r] * scale - rmax[r]) * inv);
    }
  }
}

// ---------------------------------------------------------------------------
// K5: av = attn (512x512) @ v (K=d, N=t) per (b,h); writes avT[t][h*512+c].
// Both operands staged fully async (v is [t][d] == BT directly).
// ---------------------------------------------------------------------------
__global__ __launch_bounds__(256) void gemm_av(const __bf16* __restrict__ attn,
                                               const __bf16* __restrict__ V,
                                               __bf16* __restrict__ avT) {
  const int bh = blockIdx.x;
  const int m0 = blockIdx.y * 128;   // c
  const int n0 = blockIdx.z * 128;   // t
  const __bf16* A = attn + (size_t)bh * 512 * 512;
  const __bf16* Bv = V + (size_t)bh * 1024 * 512;
  __shared__ __align__(16) __bf16 sA[2][128 * 32];
  __shared__ __align__(16) __bf16 sBT[2][128 * 32];
  const int tid = threadIdx.x, lane = tid & 31, w = tid >> 5;
  const int ar = tid >> 2, ak = (tid & 3) * 8;

  v8f acc[8];
#pragma unroll
  for (int i = 0; i < 8; ++i) acc[i] = zero_v8f();

  auto issue = [&](int buf, int k0) {
#pragma unroll
    for (int ii = 0; ii < 2; ++ii) {
      const int r = ar + ii * 64;
      copy16_g2l(&A[(size_t)(m0 + r) * 512 + k0 + ak], &sA[buf][r * 32 + ak]);
      copy16_g2l(&Bv[(size_t)(n0 + r) * 512 + k0 + ak], &sBT[buf][r * 32 + ak]);
    }
  };

  issue(0, 0);
  for (int kt = 0; kt < 16; ++kt) {
    async_wait_all();
    __syncthreads();
    const int cur = kt & 1;
    if (kt + 1 < 16) issue(cur ^ 1, (kt + 1) * 32);
    const v16bf bf = frag_b(&sBT[cur][w * 16 * 32], 32, lane);
    v16bf af = frag_a(&sA[cur][0], 32, lane);
#pragma unroll
    for (int mt = 0; mt < 8; ++mt) {
      v16bf nxt = af;
      if (mt < 7) nxt = frag_a(&sA[cur][(mt + 1) * 16 * 32], 32, lane);
      acc[mt] = wmma_bf16(af, bf, acc[mt]);
      af = nxt;
    }
  }

  const int b = bh >> 2, hd = bh & 3, hl = lane >> 4;
  const int ngl = n0 + w * 16 + (lane & 15);          // t index
#pragma unroll
  for (int mt = 0; mt < 8; ++mt) {
#pragma unroll
    for (int r = 0; r < 8; ++r) {
      const int mg = m0 + mt * 16 + 8 * hl + r;       // c index
      avT[((size_t)b * 1024 + ngl) * 2048 + hd * 512 + mg] = (__bf16)acc[mt][r];
    }
  }
}

// ---------------------------------------------------------------------------
// K6: out = avT (8192x2048) @ w_out (2048x512) + b_out + residual -> f32
// ---------------------------------------------------------------------------
__global__ __launch_bounds__(256) void gemm_out(const __bf16* __restrict__ A,
                                                const __bf16* __restrict__ B,
                                                const float* __restrict__ x,
                                                const float* __restrict__ bias,
                                                float* __restrict__ out) {
  const int row0 = blockIdx.x * 128;
  const int col0 = blockIdx.y * 128;
  __shared__ __align__(16) __bf16 sA[2][128 * 32];
  __shared__ __align__(16) __bf16 sBT[2][128 * 32];
  const int tid = threadIdx.x, lane = tid & 31, w = tid >> 5;
  const int ar = tid >> 2, ak = (tid & 3) * 8;
  const int bk = tid >> 4, bn = (tid & 15) * 8;

  v8f acc[8];
#pragma unroll
  for (int i = 0; i < 8; ++i) acc[i] = zero_v8f();

  auto issue_a = [&](int buf, int k0) {
#pragma unroll
    for (int ii = 0; ii < 2; ++ii) {
      const int r = ar + ii * 64;
      copy16_g2l(&A[(size_t)(row0 + r) * 2048 + k0 + ak], &sA[buf][r * 32 + ak]);
    }
  };
  auto load_b = [&](int k0, v8bf* br) {
#pragma unroll
    for (int ii = 0; ii < 2; ++ii)
      br[ii] = *(const v8bf*)&B[(size_t)(k0 + bk + ii * 16) * 512 + col0 + bn];
  };
  auto store_b = [&](int buf, v8bf* br) {
#pragma unroll
    for (int ii = 0; ii < 2; ++ii)
#pragma unroll
      for (int e = 0; e < 8; ++e)
        sBT[buf][(bn + e) * 32 + bk + ii * 16] = br[ii][e];
  };

  { v8bf br[2]; issue_a(0, 0); load_b(0, br); store_b(0, br); }

  for (int kt = 0; kt < 64; ++kt) {
    async_wait_all();
    __syncthreads();
    const int cur = kt & 1;
    const bool more = (kt + 1) < 64;
    v8bf br[2];
    if (more) { issue_a(cur ^ 1, (kt + 1) * 32); load_b((kt + 1) * 32, br); }
    const v16bf bf = frag_b(&sBT[cur][w * 16 * 32], 32, lane);
    v16bf af = frag_a(&sA[cur][0], 32, lane);
#pragma unroll
    for (int mt = 0; mt < 8; ++mt) {
      v16bf nxt = af;
      if (mt < 7) nxt = frag_a(&sA[cur][(mt + 1) * 16 * 32], 32, lane);
      acc[mt] = wmma_bf16(af, bf, acc[mt]);
      af = nxt;
    }
    if (more) store_b(cur ^ 1, br);
  }

  const int ngl = col0 + w * 16 + (lane & 15);
  const int hl = lane >> 4;
  const float bv = bias[ngl];
#pragma unroll
  for (int mt = 0; mt < 8; ++mt) {
#pragma unroll
    for (int r = 0; r < 8; ++r) {
      const int mg = row0 + mt * 16 + 8 * hl + r;
      const size_t idx = (size_t)mg * 512 + ngl;
      out[idx] = acc[mt][r] + x[idx] + bv;
    }
  }
}

// ---------------------------------------------------------------------------
extern "C" void kernel_launch(void* const* d_in, const int* in_sizes, int n_in,
                              void* d_out, int out_size, void* d_ws, size_t ws_size,
                              hipStream_t stream) {
  (void)in_sizes; (void)n_in; (void)out_size; (void)ws_size;
  const float* x        = (const float*)d_in[0];
  const float* gn_scale = (const float*)d_in[1];
  const float* gn_bias  = (const float*)d_in[2];
  const float* w_qkv    = (const float*)d_in[3];
  const float* b_qkv    = (const float*)d_in[4];
  const float* w_out    = (const float*)d_in[5];
  const float* b_out    = (const float*)d_in[6];
  float* out = (float*)d_out;

  char* ws = (char*)d_ws;
  size_t off = 0;
  auto alloc = [&](size_t bytes) {
    void* p = ws + off;
    off += (bytes + 255) & ~(size_t)255;
    return p;
  };
  __bf16* hbf  = (__bf16*)alloc(8192ull * 512 * 2);        // normalized x, bf16
  __bf16* wqb  = (__bf16*)alloc(512ull * 6144 * 2);        // w_qkv bf16
  __bf16* wob  = (__bf16*)alloc(2048ull * 512 * 2);        // w_out bf16
  __bf16* qT   = (__bf16*)alloc(32ull * 512 * 1024 * 2);   // [b][h][c][t]
  __bf16* kT   = (__bf16*)alloc(32ull * 512 * 1024 * 2);   // [b][h][d][t]
  __bf16* V    = (__bf16*)alloc(32ull * 1024 * 512 * 2);   // [b][h][t][d]
  __bf16* attn = (__bf16*)alloc(32ull * 512 * 512 * 2);    // [b][h][c][d]
  __bf16* avT  = (__bf16*)alloc(8192ull * 2048 * 2);       // [b*t][h*512+c]

  gn_kernel<<<dim3(256), dim3(256), 0, stream>>>(x, gn_scale, gn_bias, hbf);
  cast_f32_bf16<<<dim3((512 * 6144 + 255) / 256), dim3(256), 0, stream>>>(w_qkv, wqb, 512 * 6144);
  cast_f32_bf16<<<dim3((2048 * 512 + 255) / 256), dim3(256), 0, stream>>>(w_out, wob, 2048 * 512);
  gemm_qkv<<<dim3(64, 48), dim3(256), 0, stream>>>(hbf, wqb, b_qkv, qT, kT, V);
  attn_softmax<<<dim3(32, 8), dim3(512), 0, stream>>>(qT, kT, attn);
  gemm_av<<<dim3(32, 4, 8), dim3(256), 0, stream>>>(attn, V, avT);
  gemm_out<<<dim3(64, 4), dim3(256), 0, stream>>>(avT, wob, x, b_out, out);
}
